// PointNetSetUpConv_11450382811200
// MI455X (gfx1250) — matrix operand order repaired and checked by reading, not compiled, not run on hardware
//
#include <hip/hip_runtime.h>
#include <hip/hip_bf16.h>

// ---------------------------------------------------------------------------
// PointNetSetUpConv for MI455X (gfx1250, wave32, WMMA)
// B=4, N=8192, M=2048, K=8, F1=F2=256
// ---------------------------------------------------------------------------

#define BB 4
#define NN 8192
#define MM 2048
#define KK 8

typedef _Float16 h16;
typedef __attribute__((ext_vector_type(16))) _Float16 v16h;
typedef __attribute__((ext_vector_type(8)))  _Float16 v8h;
typedef __attribute__((ext_vector_type(8)))  float    v8f;

__device__ __forceinline__ v8f wmma_f16(v16h a, v16h b, v8f c) {
    // D = A(16x32 f16) * B(32x16 f16) + C(16x16 f32)
    return __builtin_amdgcn_wmma_f32_16x16x32_f16(
        /*neg_a=*/false, a, /*neg_b=*/false, b,
        /*c_mod=*/(short)0, c, /*reuse_a=*/false, /*reuse_b=*/false);
}

// A-fragment from row-major [row][32] LDS slab.
// Layout: lanes 0-15 row M=lane, K 0-15 (halves 0-15);
// lanes 16-31 same row, K 16-31.  kb = (lane>=16)?8:0 (K base = 2*kb).
__device__ __forceinline__ v16h load_a_frag(const h16* rowp, int kb) {
    v8h lo = *(const v8h*)(rowp + 2 * kb);
    v8h hi = *(const v8h*)(rowp + 2 * kb + 8);
    v16h a;
#pragma unroll
    for (int i = 0; i < 8; ++i) { a[i] = lo[i]; a[i + 8] = hi[i]; }
    return a;
}

// ---------------------------------------------------------------------------
// Kernel 1: convert weights to f16 (w1_0 padded 259 -> 288 with zeros)
// ---------------------------------------------------------------------------
__global__ void __launch_bounds__(256)
convert_weights_kernel(const float* __restrict__ w0, const float* __restrict__ w1,
                       const float* __restrict__ w2, const float* __restrict__ w3,
                       h16* __restrict__ w0h, h16* __restrict__ w1h,
                       h16* __restrict__ w2h, h16* __restrict__ w3h) {
    int i = blockIdx.x * 256 + threadIdx.x;
    if (i < 128 * 288) {                       // [128][288], K padded
        int o = i / 288, c = i % 288;
        w0h[i] = (c < 259) ? (h16)w0[o * 259 + c] : (h16)0.0f;
    }
    if (i < 128 * 128) w1h[i] = (h16)w1[i];    // [128][128]
    if (i < 256 * 128) w2h[i] = (h16)w2[i];    // [256][128]
    if (i < 256 * 512) w3h[i] = (h16)w3[i];    // [256][512]
}

// ---------------------------------------------------------------------------
// Kernel 2: KNN  (top-8 nearest pos2 points per pos1 point)
// grid (N/256, B), 256 threads; pos2 batch staged in LDS (24 KB)
// ---------------------------------------------------------------------------
__global__ void __launch_bounds__(256)
knn_kernel(const float* __restrict__ pos1, const float* __restrict__ pos2,
           int* __restrict__ idx) {
    __shared__ float p2[3 * MM];
    const int b = blockIdx.y;
    const int n = blockIdx.x * 256 + threadIdx.x;
    for (int i = threadIdx.x; i < 3 * MM; i += 256)
        p2[i] = pos2[b * 3 * MM + i];
    __syncthreads();

    const float px = pos1[(b * 3 + 0) * NN + n];
    const float py = pos1[(b * 3 + 1) * NN + n];
    const float pz = pos1[(b * 3 + 2) * NN + n];

    float bd[KK]; int bi[KK];
#pragma unroll
    for (int j = 0; j < KK; ++j) { bd[j] = 3.0e38f; bi[j] = 0; }

    for (int m = 0; m < MM; ++m) {
        float dx = px - p2[m];
        float dy = py - p2[MM + m];
        float dz = pz - p2[2 * MM + m];
        float d = dx * dx + dy * dy + dz * dz;
        if (d < bd[KK - 1]) {
            bd[KK - 1] = d; bi[KK - 1] = m;
#pragma unroll
            for (int j = KK - 1; j > 0; --j) {
                if (bd[j] < bd[j - 1]) {
                    float td = bd[j]; bd[j] = bd[j - 1]; bd[j - 1] = td;
                    int   ti = bi[j]; bi[j] = bi[j - 1]; bi[j - 1] = ti;
                }
            }
        }
    }
#pragma unroll
    for (int j = 0; j < KK; ++j) idx[((b * NN) + n) * KK + j] = bi[j];
}

// ---------------------------------------------------------------------------
// Kernel 3: fused gather + MLP1 (259->128->128->256) + max over K
// grid (N/16, B), 256 threads = 8 wave32.  128 columns = 16 points x 8 nbrs.
// Dynamic LDS layout (halves unless noted):
//   Wbuf  [256][32]   8192 h   (per-K-chunk weight slab)
//   Xs    [128][32]   4096 h   (layer-0 gathered input chunk, col-major K)
//   act1  [128 cols][128 ch]  16384 h
//   act2  [128 cols][128 ch]  16384 h
//   idxs  128 int, p1s 48 float
// ---------------------------------------------------------------------------
__global__ void __launch_bounds__(256)
mlp1_kernel(const float* __restrict__ pos1, const float* __restrict__ pos2,
            const float* __restrict__ feature2, const int* __restrict__ idx,
            const h16* __restrict__ w0h, const float* __restrict__ s0, const float* __restrict__ b0,
            const h16* __restrict__ w1h, const float* __restrict__ s1, const float* __restrict__ b1,
            const h16* __restrict__ w2h, const float* __restrict__ s2, const float* __restrict__ b2,
            float* __restrict__ Y) {
    extern __shared__ __attribute__((aligned(128))) char smem[];
    h16*   Wbuf = (h16*)smem;            // 8192
    h16*   Xs   = Wbuf + 8192;           // 4096
    h16*   act1 = Xs + 4096;             // 16384
    h16*   act2 = act1 + 16384;          // 16384
    int*   idxs = (int*)(act2 + 16384);  // 128
    float* p1s  = (float*)(idxs + 128);  // 48

    const int b    = blockIdx.y;
    const int n0   = blockIdx.x * 16;
    const int tid  = threadIdx.x;
    const int wave = tid >> 5;
    const int lane = tid & 31;
    const int hi   = lane >> 4;    // lane-half
    const int l16  = lane & 15;

    if (tid < 128) idxs[tid] = idx[((b * NN) + n0 + (tid >> 3)) * KK + (tid & 7)];
    if (tid < 48)  p1s[tid]  = pos1[(b * 3 + (tid >> 4)) * NN + n0 + (tid & 15)];

    const v8f vzero = {};

    // ---------------- layer 0: 128 rows, K = 288 (9 chunks) ----------------
    v8f acc[8];
#pragma unroll
    for (int t = 0; t < 8; ++t) acc[t] = vzero;

    for (int kc = 0; kc < 9; ++kc) {
        __syncthreads();
        for (int i = tid; i < 4096; i += 256) {              // weights 128x32
            int r = i >> 5, t = i & 31;
            Wbuf[i] = w0h[r * 288 + kc * 32 + t];
        }
        for (int i = tid; i < 4096; i += 256) {              // gathered input
            int t = i >> 7, c0 = i & 127;
            int c = kc * 32 + t;
            int m = idxs[c0];
            float v;
            if (c < 256)      v = feature2[(b * 256 + c) * MM + m];
            else if (c < 259) { int cc = c - 256;
                                v = pos2[(b * 3 + cc) * MM + m] - p1s[cc * 16 + (c0 >> 3)]; }
            else              v = 0.0f;
            Xs[c0 * 32 + t] = (h16)v;
        }
        __syncthreads();
        v16h a = load_a_frag(Wbuf + (wave * 16 + l16) * 32, hi * 8);
#pragma unroll
        for (int ct = 0; ct < 8; ++ct) {
            v16h bf = *(const v16h*)(Xs + (ct * 16 + l16) * 32 + hi * 16);
            acc[ct] = wmma_f16(a, bf, acc[ct]);
        }
    }
    {   // epilogue -> act1 (f16, [col][channel])
        float ss[8], bb[8];
#pragma unroll
        for (int vr = 0; vr < 8; ++vr) {
            int r = wave * 16 + hi * 8 + vr; ss[vr] = s0[r]; bb[vr] = b0[r];
        }
#pragma unroll
        for (int ct = 0; ct < 8; ++ct) {
            v8h p;
#pragma unroll
            for (int vr = 0; vr < 8; ++vr) {
                float f = acc[ct][vr] * ss[vr] + bb[vr];
                p[vr] = (h16)(f > 0.0f ? f : 0.0f);
            }
            *(v8h*)(act1 + (ct * 16 + l16) * 128 + wave * 16 + hi * 8) = p;
        }
    }

    // ---------------- layer 1: 128 rows, K = 128 (4 chunks) ----------------
#pragma unroll
    for (int t = 0; t < 8; ++t) acc[t] = vzero;
    for (int kc = 0; kc < 4; ++kc) {
        __syncthreads();
        for (int i = tid; i < 4096; i += 256) {
            int r = i >> 5, t = i & 31;
            Wbuf[i] = w1h[r * 128 + kc * 32 + t];
        }
        __syncthreads();
        v16h a = load_a_frag(Wbuf + (wave * 16 + l16) * 32, hi * 8);
#pragma unroll
        for (int ct = 0; ct < 8; ++ct) {
            v16h bf = *(const v16h*)(act1 + (ct * 16 + l16) * 128 + kc * 32 + hi * 16);
            acc[ct] = wmma_f16(a, bf, acc[ct]);
        }
    }
    {   // epilogue -> act2
        float ss[8], bb[8];
#pragma unroll
        for (int vr = 0; vr < 8; ++vr) {
            int r = wave * 16 + hi * 8 + vr; ss[vr] = s1[r]; bb[vr] = b1[r];
        }
#pragma unroll
        for (int ct = 0; ct < 8; ++ct) {
            v8h p;
#pragma unroll
            for (int vr = 0; vr < 8; ++vr) {
                float f = acc[ct][vr] * ss[vr] + bb[vr];
                p[vr] = (h16)(f > 0.0f ? f : 0.0f);
            }
            *(v8h*)(act2 + (ct * 16 + l16) * 128 + wave * 16 + hi * 8) = p;
        }
    }

    // ---------------- layer 2: 256 rows, K = 128 (4 chunks) ----------------
    v8f acc2[16];
#pragma unroll
    for (int t = 0; t < 16; ++t) acc2[t] = vzero;
    for (int kc = 0; kc < 4; ++kc) {
        __syncthreads();
        for (int i = tid; i < 8192; i += 256) {
            int r = i >> 5, t = i & 31;
            Wbuf[i] = w2h[r * 128 + kc * 32 + t];
        }
        __syncthreads();
        v16h a0 = load_a_frag(Wbuf + (wave * 32 + l16) * 32, hi * 8);
        v16h a1 = load_a_frag(Wbuf + (wave * 32 + 16 + l16) * 32, hi * 8);
#pragma unroll
        for (int ct = 0; ct < 8; ++ct) {
            v16h bf = *(const v16h*)(act2 + (ct * 16 + l16) * 128 + kc * 32 + hi * 16);
            acc2[ct]     = wmma_f16(a0, bf, acc2[ct]);
            acc2[8 + ct] = wmma_f16(a1, bf, acc2[8 + ct]);
        }
    }
    // epilogue: scale/bias/ReLU, max over 8 neighbors (8-lane groups), write Y
#pragma unroll
    for (int rt = 0; rt < 2; ++rt) {
        float ss[8], bb[8];
#pragma unroll
        for (int vr = 0; vr < 8; ++vr) {
            int r = wave * 32 + rt * 16 + hi * 8 + vr; ss[vr] = s2[r]; bb[vr] = b2[r];
        }
#pragma unroll
        for (int ct = 0; ct < 8; ++ct) {
#pragma unroll
            for (int vr = 0; vr < 8; ++vr) {
                float f = acc2[rt * 8 + ct][vr] * ss[vr] + bb[vr];
                f = f > 0.0f ? f : 0.0f;
                f = fmaxf(f, __shfl_xor(f, 1, 32));
                f = fmaxf(f, __shfl_xor(f, 2, 32));
                f = fmaxf(f, __shfl_xor(f, 4, 32));
                if ((lane & 7) == 0) {
                    int p = ct * 2 + ((lane >> 3) & 1);
                    int r = wave * 32 + rt * 16 + hi * 8 + vr;
                    Y[((b * 256) + r) * NN + n0 + p] = f;
                }
            }
        }
    }
}

// ---------------------------------------------------------------------------
// Kernel 4: MLP2  out[256 x N] = relu(s*(W[256x512] @ concat(Y, feature1)) + b)
// grid (N/128, B), 256 threads = 8 wave32, 128-column tiles, K = 512.
// ---------------------------------------------------------------------------
__global__ void __launch_bounds__(256)
mlp2_kernel(const float* __restrict__ Y, const float* __restrict__ feature1,
            const h16* __restrict__ w3h, const float* __restrict__ s,
            const float* __restrict__ bias, float* __restrict__ out) {
    __shared__ __attribute__((aligned(128))) h16 Wbuf[8192];   // [256][32]
    __shared__ __attribute__((aligned(128))) h16 Xs[4096];     // [128][32]

    const int b    = blockIdx.y;
    const int n0   = blockIdx.x * 128;
    const int tid  = threadIdx.x;
    const int wave = tid >> 5;
    const int lane = tid & 31;
    const int hi   = lane >> 4;
    const int l16  = lane & 15;

    const v8f vzero = {};
    v8f acc[16];
#pragma unroll
    for (int t = 0; t < 16; ++t) acc[t] = vzero;

    for (int kc = 0; kc < 16; ++kc) {
        __syncthreads();
        for (int i = tid; i < 8192; i += 256) {
            int r = i >> 5, t = i & 31;
            Wbuf[i] = w3h[r * 512 + kc * 32 + t];
        }
        for (int i = tid; i < 4096; i += 256) {
            int t = i >> 7, c0 = i & 127;
            int c = kc * 32 + t;
            float v = (c < 256) ? Y[((b * 256) + c) * NN + n0 + c0]
                                : feature1[((b * 256) + (c - 256)) * NN + n0 + c0];
            Xs[c0 * 32 + t] = (h16)v;
        }
        __syncthreads();
        v16h a0 = load_a_frag(Wbuf + (wave * 32 + l16) * 32, hi * 8);
        v16h a1 = load_a_frag(Wbuf + (wave * 32 + 16 + l16) * 32, hi * 8);
#pragma unroll
        for (int ct = 0; ct < 8; ++ct) {
            v16h bf = *(const v16h*)(Xs + (ct * 16 + l16) * 32 + hi * 16);
            acc[ct]     = wmma_f16(a0, bf, acc[ct]);
            acc[8 + ct] = wmma_f16(a1, bf, acc[8 + ct]);
        }
    }
#pragma unroll
    for (int rt = 0; rt < 2; ++rt) {
        float ss[8], bb[8];
#pragma unroll
        for (int vr = 0; vr < 8; ++vr) {
            int r = wave * 32 + rt * 16 + hi * 8 + vr; ss[vr] = s[r]; bb[vr] = bias[r];
        }
#pragma unroll
        for (int ct = 0; ct < 8; ++ct) {
            int n = n0 + ct * 16 + l16;
#pragma unroll
            for (int vr = 0; vr < 8; ++vr) {
                int r = wave * 32 + rt * 16 + hi * 8 + vr;
                float f = acc[rt * 8 + ct][vr] * ss[vr] + bb[vr];
                out[((b * 256) + r) * NN + n] = f > 0.0f ? f : 0.0f;
            }
        }
    }
}

// ---------------------------------------------------------------------------
// Host launcher
// ---------------------------------------------------------------------------
extern "C" void kernel_launch(void* const* d_in, const int* in_sizes, int n_in,
                              void* d_out, int out_size, void* d_ws, size_t ws_size,
                              hipStream_t stream) {
    const float* pos1     = (const float*)d_in[0];
    const float* pos2     = (const float*)d_in[1];
    const float* feature1 = (const float*)d_in[2];
    const float* feature2 = (const float*)d_in[3];
    const float* w1_0 = (const float*)d_in[4];
    const float* s1_0 = (const float*)d_in[5];
    const float* b1_0 = (const float*)d_in[6];
    const float* w1_1 = (const float*)d_in[7];
    const float* s1_1 = (const float*)d_in[8];
    const float* b1_1 = (const float*)d_in[9];
    const float* w1_2 = (const float*)d_in[10];
    const float* s1_2 = (const float*)d_in[11];
    const float* b1_2 = (const float*)d_in[12];
    const float* w2_0 = (const float*)d_in[13];
    const float* s2_0 = (const float*)d_in[14];
    const float* b2_0 = (const float*)d_in[15];

    // Workspace layout (bytes)
    char* ws = (char*)d_ws;
    int*  idx = (int*)ws;                                  //  4*8192*8*4 = 1 MiB
    size_t off = (size_t)BB * NN * KK * sizeof(int);
    h16* w0h = (h16*)(ws + off); off += 128 * 288 * sizeof(h16);
    h16* w1h = (h16*)(ws + off); off += 128 * 128 * sizeof(h16);
    h16* w2h = (h16*)(ws + off); off += 256 * 128 * sizeof(h16);
    h16* w3h = (h16*)(ws + off); off += 256 * 512 * sizeof(h16);
    off = (off + 255) & ~(size_t)255;
    float* Y = (float*)(ws + off);                         // 4*256*8192*4 = 32 MiB

    convert_weights_kernel<<<(256 * 512 + 255) / 256, 256, 0, stream>>>(
        w1_0, w1_1, w1_2, w2_0, w0h, w1h, w2h, w3h);

    knn_kernel<<<dim3(NN / 256, BB), 256, 0, stream>>>(pos1, pos2, idx);

    const size_t smem1 = 8192 * sizeof(h16) + 4096 * sizeof(h16) +
                         2 * 16384 * sizeof(h16) + 128 * sizeof(int) + 48 * sizeof(float);
    (void)hipFuncSetAttribute((const void*)mlp1_kernel,
                              hipFuncAttributeMaxDynamicSharedMemorySize, (int)smem1);
    mlp1_kernel<<<dim3(NN / 16, BB), 256, smem1, stream>>>(
        pos1, pos2, feature2, idx,
        w0h, s1_0, b1_0, w1h, s1_1, b1_1, w2h, s1_2, b1_2, Y);

    mlp2_kernel<<<dim3(NN / 128, BB), 256, 0, stream>>>(
        Y, feature1, w3h, s2_0, b2_0, (float*)d_out);
}
